// NuclearConsistencyLoss_40862318854227
// MI455X (gfx1250) — matrix operand order repaired and checked by reading, not compiled, not run on hardware
//
#include <hip/hip_runtime.h>

// Problem geometry (fixed by the reference).
#define NB   16
#define CH   3
#define HI   512
#define WI   512
#define NPIX (NB * HI * WI)        // 4,194,304 pixels per single-channel field
#define TS   16                    // WMMA tile side
#define NTX  (WI / TS)             // 32
#define NTY  (HI / TS)             // 32
#define NPART 4096

typedef __attribute__((ext_vector_type(16))) _Float16 v16h;
typedef __attribute__((ext_vector_type(8)))  float    v8f;

// DT kernel weights: exp(-hypot(dx,dy)/0.35)
#define W_E1 0.05743156f   // exp(-1/0.35)
#define W_E2 0.01758670f   // exp(-sqrt(2)/0.35)
#define DT_HH 0.35f

// ---------------------------------------------------------------------------
// 1) Horizontal 7-tap max of dapi = target[:,0] (clipped window == SAME w/ -inf)
__global__ void k_rowmax(const float* __restrict__ target, float* __restrict__ tmp) {
    int gid = blockIdx.x * blockDim.x + threadIdx.x;
    if (gid >= NPIX) return;
    int x = gid % WI, y = (gid / WI) % HI, n = gid / (WI * HI);
    const float* dapi = target + (size_t)n * CH * HI * WI;   // channel 0
    int x0 = max(x - 3, 0), x1 = min(x + 3, WI - 1);
    float m = dapi[(size_t)y * WI + x0];
    for (int xx = x0 + 1; xx <= x1; ++xx) m = fmaxf(m, dapi[(size_t)y * WI + xx]);
    tmp[gid] = m;
}

// 2) Vertical 7-tap max, then outside / binary / dist init / global binary count.
__global__ void k_prep(const float* __restrict__ tmp, float* __restrict__ outside,
                       float* __restrict__ b0, float* __restrict__ dist,
                       int* __restrict__ cnt) {
    int gid = blockIdx.x * blockDim.x + threadIdx.x;
    if (gid >= NPIX) return;
    int x = gid % WI, y = (gid / WI) % HI, n = gid / (WI * HI);
    const float* t = tmp + (size_t)n * HI * WI;
    int y0 = max(y - 3, 0), y1 = min(y + 3, HI - 1);
    float m = t[(size_t)y0 * WI + x];
    for (int yy = y0 + 1; yy <= y1; ++yy) m = fmaxf(m, t[(size_t)yy * WI + x]);
    // dilate(sigmoid(k*(d-th))) == sigmoid(k*(dilate(d)-th)) by monotonicity
    float sg = 1.0f / (1.0f + expf(-10.0f * (m - 0.97f)));
    outside[gid] = 1.0f - sg;
    int isb = (m > 0.97f) ? 1 : 0;                 // dilated > 0.5  <=>  m > 0.97
    b0[gid]   = (float)isb;
    dist[gid] = 0.0f;
    unsigned long long bal = __ballot(isb != 0);   // wave32: low 32 bits
    if ((threadIdx.x & 31) == 0) atomicAdd(cnt, __popcll(bal));
}

__global__ void k_init(int* cnt) {
    if (threadIdx.x == 0 && blockIdx.x == 0) *cnt = 0;
}

// ---------------------------------------------------------------------------
// 3) One distance-transform iteration. One wave32 per 16x16 tile.
//    conv = P0 + shift(P1,-1) + shift(P1,+1) where P0/P1 are vertical 3-tap
//    convs done as banded-matrix WMMAs:  P = A_band(16x32 f16) x B_patch(32x16 f16).
__global__ __launch_bounds__(32) void k_dt(const float* __restrict__ bin,
                                           float* __restrict__ bout,
                                           float* __restrict__ dist, int iter) {
    const int lane = threadIdx.x;
    const int half = lane >> 4;          // 0: lanes 0-15, 1: lanes 16-31
    const int l    = lane & 15;
    int tid = blockIdx.x;
    int tx = tid % NTX, ty = (tid / NTX) % NTY, n = tid / (NTX * NTY);
    const int rb = ty * TS, cb = tx * TS;
    const float* bimg = bin + (size_t)n * HI * WI;

    // Build banded A matrices and the B image patch in the ISA VGPR layouts.
    // A (16x32 f16): lane element j holds K = j + (j>=8 ? 8:0) + half*8, row M = l.
    // B (32x16 f16): lane element j holds K = j + (j>=8 ? 8:0) + half*8, col N = l.
    v16h a0, a1, bm;
#pragma unroll
    for (int j = 0; j < 16; ++j) {
        int K = j + ((j >= 8) ? 8 : 0) + (half ? 8 : 0);
        int d = K - l;                       // band: output row r=l uses K=r..r+2
        float w0 = 0.0f, w1 = 0.0f;
        if (d == 1)              { w0 = 1.0f;  w1 = W_E1; }
        else if (d == 0 || d == 2) { w0 = W_E1; w1 = W_E2; }
        a0[j] = (_Float16)w0;
        a1[j] = (_Float16)w1;
        int kb = j + ((j >= 8) ? 8 : 0) + half * 8;   // B-patch row index
        float bv = 0.0f;
        if (kb < TS + 2) {                   // 18 live rows: rb-1 .. rb+16 (replicated)
            int ry = rb - 1 + kb;
            ry = min(max(ry, 0), HI - 1);
            bv = bimg[(size_t)ry * WI + (cb + l)];
        }
        bm[j] = (_Float16)bv;
    }
    v8f zero = {0.f, 0.f, 0.f, 0.f, 0.f, 0.f, 0.f, 0.f};
    v8f p0 = __builtin_amdgcn_wmma_f32_16x16x32_f16(false, a0, false, bm,
                                                    (short)0, zero, false, false);
    v8f p1 = __builtin_amdgcn_wmma_f32_16x16x32_f16(false, a1, false, bm,
                                                    (short)0, zero, false, false);

    // Tile-edge columns of P1 (x = cb-1 and cb+16), computed by the edge lanes.
    float bl[8] = {0}, br[8] = {0};
    if (l == 0) {
        int xl = max(cb - 1, 0);
#pragma unroll
        for (int j = 0; j < 8; ++j) {
            int r = j + half * 8;
            float s = 0.0f;
            for (int dy = -1; dy <= 1; ++dy) {
                int ry = min(max(rb + r + dy, 0), HI - 1);
                s += ((dy == 0) ? W_E1 : W_E2) * bimg[(size_t)ry * WI + xl];
            }
            bl[j] = s;
        }
    }
    if (l == 15) {
        int xr = min(cb + TS, WI - 1);
#pragma unroll
        for (int j = 0; j < 8; ++j) {
            int r = j + half * 8;
            float s = 0.0f;
            for (int dy = -1; dy <= 1; ++dy) {
                int ry = min(max(rb + r + dy, 0), HI - 1);
                s += ((dy == 0) ? W_E1 : W_E2) * bimg[(size_t)ry * WI + xr];
            }
            br[j] = s;
        }
    }

    // D layout: VGPR j -> row M = j + half*8, col N = l. Combine + DT update.
#pragma unroll
    for (int j = 0; j < 8; ++j) {
        float left  = __shfl(p1[j], lane - 1, 32);
        float right = __shfl(p1[j], lane + 1, 32);
        if (l == 0)  left  = bl[j];
        if (l == 15) right = br[j];
        float v = p0[j] + left + right;                 // 3x3 weighted conv
        int y = rb + half * 8 + j, x = cb + l;
        size_t idx = (size_t)n * HI * WI + (size_t)y * WI + x;
        float bold = bimg[(size_t)y * WI + x];
        // cdt = -h*log(conv); conv==0 -> +inf -> nan_to_num -> 0; mask = cdt>0
        bool mask = (v > 0.0f) && (v < 1.0f);
        if (mask) {
            float cdt = -DT_HH * logf(v);
            dist[idx] += (float)iter + cdt;             // r = 1
            bout[idx] = 1.0f;
        } else {
            bout[idx] = bold;
        }
    }
}

// ---------------------------------------------------------------------------
// 4) Deterministic two-stage mean of lap2 * outside * penalty.
__global__ void k_partial(const float* __restrict__ x1hat,
                          const float* __restrict__ outside,
                          const float* __restrict__ dist,
                          const int* __restrict__ cnt,
                          float* __restrict__ partial) {
    __shared__ float sm[256];
    int tid = threadIdx.x;
    int gid = blockIdx.x * blockDim.x + tid;
    int stride = gridDim.x * blockDim.x;
    bool hasb = (*cnt) > 0;
    float acc = 0.0f;
    for (int i = gid; i < NPIX; i += stride) {
        int x = i % WI, y = (i / WI) % HI, n = i / (WI * HI);
        float lap2 = x1hat[((size_t)n * CH + 1) * HI * WI + (size_t)y * WI + x];
        float pen = hasb ? fminf(fmaxf(dist[i] / 10.0f, 0.0f), 1.0f) : 1.0f;
        acc += lap2 * outside[i] * pen;
    }
    sm[tid] = acc;
    __syncthreads();
    for (int s = 128; s > 0; s >>= 1) {
        if (tid < s) sm[tid] += sm[tid + s];
        __syncthreads();
    }
    if (tid == 0) partial[blockIdx.x] = sm[0];
}

__global__ void k_final(const float* __restrict__ partial, float* __restrict__ out) {
    __shared__ float sm[256];
    int tid = threadIdx.x;
    float acc = 0.0f;
    for (int i = tid; i < NPART; i += 256) acc += partial[i];
    sm[tid] = acc;
    __syncthreads();
    for (int s = 128; s > 0; s >>= 1) {
        if (tid < s) sm[tid] += sm[tid + s];
        __syncthreads();
    }
    if (tid == 0) out[0] = sm[0] / (float)NPIX;
}

// ---------------------------------------------------------------------------
extern "C" void kernel_launch(void* const* d_in, const int* in_sizes, int n_in,
                              void* d_out, int out_size, void* d_ws, size_t ws_size,
                              hipStream_t stream) {
    (void)in_sizes; (void)n_in; (void)out_size; (void)ws_size;
    const float* x1hat  = (const float*)d_in[0];
    const float* target = (const float*)d_in[1];
    float* out = (float*)d_out;

    char* ws = (char*)d_ws;
    const size_t fB = (size_t)NPIX * sizeof(float);
    float* outside = (float*)(ws + 0 * fB);
    float* bA      = (float*)(ws + 1 * fB);
    float* bB      = (float*)(ws + 2 * fB);   // also reused as row-max temp pre-DT
    float* dist    = (float*)(ws + 3 * fB);
    float* partial = (float*)(ws + 4 * fB);
    int*   cnt     = (int*)(ws + 4 * fB + NPART * sizeof(float));

    dim3 blk(256), grd(NPIX / 256);
    k_init<<<dim3(1), dim3(32), 0, stream>>>(cnt);
    k_rowmax<<<grd, blk, 0, stream>>>(target, bB);
    k_prep<<<grd, blk, 0, stream>>>(bB, outside, bA, dist, cnt);

    // DT: n_iters = ceil(512/1) = 512; extra post-convergence iterations are
    // idempotent (mask==0 everywhere), matching the while_loop early exit.
    const int ntiles = NTX * NTY * NB;   // 16384 wave32 tiles
    for (int it = 0; it < 512; ++it) {
        const float* bi = (it & 1) ? bB : bA;
        float*       bo = (it & 1) ? bA : bB;
        k_dt<<<dim3(ntiles), dim3(32), 0, stream>>>(bi, bo, dist, it);
    }

    k_partial<<<dim3(NPART), blk, 0, stream>>>(x1hat, outside, dist, cnt, partial);
    k_final<<<dim3(1), blk, 0, stream>>>(partial, out);
}
